// MultiHeadedAttention_17360257810690
// MI455X (gfx1250) — compile-verified, matrix operand order checked by reading, not run on hardware
//
#include <hip/hip_runtime.h>

#define DEVINL __device__ __forceinline__

typedef __bf16 bf16_t;
typedef __attribute__((ext_vector_type(16))) __bf16 bf16x16;
typedef __attribute__((ext_vector_type(8)))  float  f32x8;

static constexpr int B_ = 2, L_ = 2048, D_ = 1024, H_ = 16, DH_ = 64;
static constexpr int BL_ = B_ * L_;
static constexpr int NC_ = L_ / 32;              // 64 key chunks of 32

// ---- bf16 helpers (round-to-nearest-even) ----
DEVINL unsigned short bfbits(float f) {
  unsigned u = __builtin_bit_cast(unsigned, f);
  unsigned r = u + 0x7FFFu + ((u >> 16) & 1u);
  return (unsigned short)(r >> 16);
}
DEVINL bf16_t tobf(float f) {
  unsigned short s = bfbits(f);
  return __builtin_bit_cast(bf16_t, s);
}
DEVINL unsigned pack2(float a, float b) {
  return (unsigned)bfbits(a) | ((unsigned)bfbits(b) << 16);
}

union OpU  { bf16x16 v; unsigned u[8]; };
union OpU4 { bf16x16 v; uint4 q[2]; };

// WMMA 16-bit operand from a row-major matrix (global).
// CDNA5 16-bit 16x32 operand layout:
//   lanes 0-15 : regs0..3 = K 0..7 (2/dword), regs4..7 = K 16..23
//   lanes 16-31: regs0..3 = K 8..15,          regs4..7 = K 24..31
DEVINL bf16x16 load_op_row(const bf16_t* base, int ld, int row, int k0, int hf) {
  OpU4 r;
  const uint4* p = (const uint4*)(base + (size_t)row * ld + k0 + 8 * hf);
  r.q[0] = p[0];
  r.q[1] = p[2];
  return r.v;
}
// Same pattern from LDS (2x ds_load_b128).
DEVINL bf16x16 lds_op_row(const unsigned short* base, int ld, int row, int k0, int hf) {
  OpU4 r;
  const uint4* p = (const uint4*)(base + row * ld + k0 + 8 * hf);
  r.q[0] = p[0];
  r.q[1] = p[2];
  return r.v;
}

DEVINL f32x8 wmma_bf16(bf16x16 a, bf16x16 b, f32x8 c) {
  return __builtin_amdgcn_wmma_f32_16x16x32_bf16(
      false, a, false, b, (short)0, c, false, false);
}

// ---- CDNA5 async global->LDS copy (ASYNCcnt path), 16B per lane ----
DEVINL unsigned lds_off32(const void* p) {       // generic LDS ptr -> 32-bit LDS addr
  return (unsigned)(uintptr_t)p;
}
DEVINL void async_copy_b128(unsigned lds_addr, const void* gptr) {
  asm volatile("global_load_async_to_lds_b128 %0, %1, off"
               :: "v"(lds_addr), "v"((unsigned long long)(uintptr_t)gptr)
               : "memory");
}
DEVINL void wait_async_le0() { asm volatile("s_wait_asynccnt 0x0" ::: "memory"); }
DEVINL void wait_async_le1() { asm volatile("s_wait_asynccnt 0x1" ::: "memory"); }
DEVINL void wait_async_le2() { asm volatile("s_wait_asynccnt 0x2" ::: "memory"); }

// ---------------- f32 -> bf16 conversion ----------------
__global__ void k_cvt(const float* __restrict__ in, bf16_t* __restrict__ out, int n) {
  int i = blockIdx.x * blockDim.x + threadIdx.x;
  int stride = gridDim.x * blockDim.x;
  for (; i < n; i += stride) out[i] = tobf(in[i]);
}

// ---------------- normalize head-0 attention rows ----------------
__global__ void k_fix(float* __restrict__ top, const float* __restrict__ invRow, int n) {
  int i = blockIdx.x * blockDim.x + threadIdx.x;
  int stride = gridDim.x * blockDim.x;
  for (; i < n; i += stride) top[i] *= invRow[i >> 11];   // row = i / L_
}

// ---------------- C = A(MxK) * W(NxK)^T, bf16 in / f32 acc ----------------
// MODE 0: f32 row-major -> outF
// MODE 1: bf16 head-major (B,H,L,DH) -> outH, scaled          (Q,K)
// MODE 2: bf16 transposed head-major (B,H,DH,L) -> outH       (V^T)
template <int MODE>
__global__ void __launch_bounds__(256)
k_gemm_xt(const bf16_t* __restrict__ A, const bf16_t* __restrict__ W,
          float* __restrict__ outF, bf16_t* __restrict__ outH, float scale) {
  const int M = BL_, N = D_, K = D_;
  const int lane = threadIdx.x & 31;
  const int wave = threadIdx.x >> 5;
  const int t = blockIdx.x * (blockDim.x >> 5) + wave;
  const int tilesN = N >> 6;
  if (t >= (M >> 4) * tilesN) return;
  const int m0 = (t / tilesN) << 4;
  const int n0 = (t % tilesN) << 6;
  const int li = lane & 15;
  const int hf = lane >> 4;

  f32x8 acc[4] = {};
  for (int k0 = 0; k0 < K; k0 += 32) {
    bf16x16 a = load_op_row(A, K, m0 + li, k0, hf);
#pragma unroll
    for (int j = 0; j < 4; ++j) {
      bf16x16 w = load_op_row(W, K, n0 + j * 16 + li, k0, hf);
      acc[j] = wmma_bf16(a, w, acc[j]);
    }
  }

#pragma unroll
  for (int j = 0; j < 4; ++j) {
    const int n = n0 + j * 16 + li;              // C-layout: lane holds column n
#pragma unroll
    for (int r = 0; r < 8; ++r) {
      const int m = m0 + r + 8 * hf;             // reg r holds row m
      float v = acc[j][r] * scale;
      const int b = m >> 11, l = m & (L_ - 1);
      const int h = n >> 6, dd = n & (DH_ - 1);
      if (MODE == 0) {
        outF[(size_t)m * N + n] = v;
      } else if (MODE == 1) {
        outH[(((size_t)b * H_ + h) * L_ + l) * DH_ + dd] = tobf(v);
      } else {
        outH[(((size_t)b * H_ + h) * DH_ + dd) * L_ + l] = tobf(v);
      }
    }
  }
}

// ---------------- attention ----------------
// Block = 8 waves sharing one (b,h); each wave owns TWO 16-query tiles so every
// staged K / V^T operand feeds two WMMAs. K (32x64) and V^T (64x32) chunks are
// double-buffered in LDS via async global->LDS copies. S^T = K*Q^T so each lane
// owns a query column; context is computed transposed: C^T = V^T * P^T.
// Pass 1 computes only the UNMASKED row max (softmax is shift-invariant, and
// unmasked-max >= masked-max keeps exp() in range) -> no mask loads, no exp.
// Pass 2 accumulates unnormalized p=exp(s-m) (<=1, bf16-safe), its sum, and the
// context; normalization is applied to the accumulator at the end, and head-0
// attention rows are rescaled by a follow-up kernel via a per-row 1/l table.
__global__ void __launch_bounds__(256)
k_attn(const bf16_t* __restrict__ Qh, const bf16_t* __restrict__ Kh,
       const bf16_t* __restrict__ Vt, const unsigned char* __restrict__ mask,
       bf16_t* __restrict__ Ctx, float* __restrict__ topAttn,
       float* __restrict__ invRow) {
  __shared__ __align__(16) unsigned short ldsK[2][32 * DH_];  // 2 x 4 KB
  __shared__ __align__(16) unsigned short ldsV[2][DH_ * 32];  // 2 x 4 KB

  const int tid  = threadIdx.x;
  const int lane = tid & 31;
  const int wave = tid >> 5;

  const int GQT = (L_ >> 4) >> 4;              // 8 groups of 16 q-tiles
  const int grp = blockIdx.x % GQT;
  const int h   = (blockIdx.x / GQT) % H_;
  const int b   = blockIdx.x / (GQT * H_);

  const int hf = lane >> 4;
  const int li = lane & 15;
  const size_t bh = ((size_t)b * H_ + h) * (size_t)L_ * DH_;
  const bf16_t* Qp  = Qh + bh;
  const bf16_t* Kp  = Kh + bh;
  const bf16_t* Vtp = Vt + bh;                 // (DH, L) slice

  // two query tiles per wave
  int qq[2];
  const unsigned char* mrow[2];
  bf16x16 bq0[2], bq1[2];
#pragma unroll
  for (int qi = 0; qi < 2; ++qi) {
    const int qt = (grp << 4) + (qi << 3) + wave;
    qq[qi] = (qt << 4) + li;
    mrow[qi] = mask + ((size_t)b * L_ + qq[qi]) * L_;
    bq0[qi] = load_op_row(Qp, DH_, qq[qi], 0, hf);
    bq1[qi] = load_op_row(Qp, DH_, qq[qi], 32, hf);
  }

  // async-copy lane assignments (one b128 per thread per tensor per chunk)
  const int krow = tid >> 3, kseg = tid & 7;   // K : 32 rows x 8 segs
  const int vrow = tid >> 2, vseg = tid & 3;   // Vt: 64 rows x 4 segs
  unsigned ldsKoff[2], ldsVoff[2];
#pragma unroll
  for (int c = 0; c < 2; ++c) {
    ldsKoff[c] = lds_off32(&ldsK[c][krow * DH_ + kseg * 8]);
    ldsVoff[c] = lds_off32(&ldsV[c][vrow * 32 + vseg * 8]);
  }

  // ---------- pass 1: per-query raw-score max (no mask, no exp) ----------
  float m_run[2] = {-1e30f, -1e30f};
  async_copy_b128(ldsKoff[0], Kp + (size_t)krow * DH_ + kseg * 8);
  for (int i = 0; i < NC_; ++i) {
    const int cur = i & 1, kc = i * 32;
    if (i + 1 < NC_) {        // prefetch next chunk into the other buffer
      async_copy_b128(ldsKoff[cur ^ 1],
                      Kp + (size_t)(kc + 32 + krow) * DH_ + kseg * 8);
      wait_async_le1();       // current chunk complete (in-order retirement)
    } else {
      wait_async_le0();
    }
    __syncthreads();
#pragma unroll
    for (int sub = 0; sub < 2; ++sub) {
      const int kb = sub << 4;
      bf16x16 a0 = lds_op_row(ldsK[cur], DH_, kb + li, 0, hf);
      bf16x16 a1 = lds_op_row(ldsK[cur], DH_, kb + li, 32, hf);
#pragma unroll
      for (int qi = 0; qi < 2; ++qi) {
        f32x8 s = {};
        s = wmma_bf16(a0, bq0[qi], s);
        s = wmma_bf16(a1, bq1[qi], s);
        float t0 = fmaxf(s[0], s[1]), t1 = fmaxf(s[2], s[3]);
        float t2 = fmaxf(s[4], s[5]), t3 = fmaxf(s[6], s[7]);
        m_run[qi] = fmaxf(m_run[qi], fmaxf(fmaxf(t0, t1), fmaxf(t2, t3)));
      }
    }
    __syncthreads();
  }
  // merge half-wave partials (same query, disjoint key rows)
  float mF[2];
#pragma unroll
  for (int qi = 0; qi < 2; ++qi)
    mF[qi] = fmaxf(m_run[qi], __shfl_xor(m_run[qi], 16, 32));

  // ---------- pass 2: p=exp(s-m), sum, C^T += V^T * P^T (unnormalized) -----
  f32x8 acc[2][4] = {};
  float l_run[2] = {0.f, 0.f};
  const bool dotop = (h == 0);
  float* trow[2];
#pragma unroll
  for (int qi = 0; qi < 2; ++qi)
    trow[qi] = topAttn + ((size_t)b * L_ + qq[qi]) * L_;

  async_copy_b128(ldsKoff[0], Kp + (size_t)krow * DH_ + kseg * 8);
  async_copy_b128(ldsVoff[0], Vtp + (size_t)vrow * L_ + vseg * 8);
  for (int i = 0; i < NC_; ++i) {
    const int cur = i & 1, kc = i * 32;
    if (i + 1 < NC_) {
      async_copy_b128(ldsKoff[cur ^ 1],
                      Kp + (size_t)(kc + 32 + krow) * DH_ + kseg * 8);
      async_copy_b128(ldsVoff[cur ^ 1],
                      Vtp + (size_t)vrow * L_ + kc + 32 + vseg * 8);
      wait_async_le2();
    } else {
      wait_async_le0();
    }
    __syncthreads();

    OpU pb[2];
    float pv[2][16];
#pragma unroll
    for (int sub = 0; sub < 2; ++sub) {
      const int kb = sub << 4;
      bf16x16 a0 = lds_op_row(ldsK[cur], DH_, kb + li, 0, hf);
      bf16x16 a1 = lds_op_row(ldsK[cur], DH_, kb + li, 32, hf);
#pragma unroll
      for (int qi = 0; qi < 2; ++qi) {
        f32x8 s = {};
        s = wmma_bf16(a0, bq0[qi], s);
        s = wmma_bf16(a1, bq1[qi], s);
        const uint2 mb = *(const uint2*)(mrow[qi] + kc + kb + 8 * hf);
        float psum = 0.f;
#pragma unroll
        for (int r = 0; r < 8; ++r) {
          const unsigned mbyte = ((r < 4 ? mb.x : mb.y) >> (8 * (r & 3))) & 0xFFu;
          const float x = mbyte ? -1e18f : s[r];
          const float pr = __expf(x - mF[qi]);   // unnormalized
          pv[qi][kb / 2 + r] = pr;               // [sub*8 + r]
          psum += pr;
          if (dotop) trow[qi][kc + kb + r + 8 * hf] = pr;
        }
        l_run[qi] += psum;
      }
    }
    // Repack P^T as B-operand: this lane's key set matches the operand swizzle
    // exactly -> pure in-lane packing, no shuffles.
#pragma unroll
    for (int qi = 0; qi < 2; ++qi) {
#pragma unroll
      for (int v = 0; v < 4; ++v)
        pb[qi].u[v] = pack2(pv[qi][2 * v], pv[qi][2 * v + 1]);
#pragma unroll
      for (int v = 0; v < 4; ++v)
        pb[qi].u[4 + v] = pack2(pv[qi][8 + 2 * v], pv[qi][8 + 2 * v + 1]);
    }

#pragma unroll
    for (int ds = 0; ds < 4; ++ds) {
      bf16x16 av = lds_op_row(ldsV[cur], 32, (ds << 4) + li, 0, hf);  // rows = d
#pragma unroll
      for (int qi = 0; qi < 2; ++qi)
        acc[qi][ds] = wmma_bf16(av, pb[qi].v, acc[qi][ds]);
    }
    __syncthreads();
  }

  // merge sums, normalize accumulator, publish per-row 1/l for the fix-up
  float inv[2];
#pragma unroll
  for (int qi = 0; qi < 2; ++qi) {
    const float lF = l_run[qi] + __shfl_xor(l_run[qi], 16, 32);
    inv[qi] = 1.0f / fmaxf(lF, 1e-30f);
    if (dotop && hf == 0) invRow[(size_t)b * L_ + qq[qi]] = inv[qi];
  }

  // C^T layout: lane holds query column q; reg r holds d = 16*ds + r + 8*hf
#pragma unroll
  for (int qi = 0; qi < 2; ++qi) {
    bf16_t* cp = Ctx + ((size_t)b * L_ + qq[qi]) * D_ + h * DH_;
#pragma unroll
    for (int ds = 0; ds < 4; ++ds)
#pragma unroll
      for (int r = 0; r < 8; ++r)
        cp[(ds << 4) + r + 8 * hf] = tobf(acc[qi][ds][r] * inv[qi]);
  }
}

// ---------------- host side ----------------
extern "C" void kernel_launch(void* const* d_in, const int* in_sizes, int n_in,
                              void* d_out, int out_size, void* d_ws, size_t ws_size,
                              hipStream_t stream) {
  (void)in_sizes; (void)n_in; (void)out_size; (void)ws_size;
  const float* q  = (const float*)d_in[0];
  const float* k  = (const float*)d_in[1];
  const float* v  = (const float*)d_in[2];
  const unsigned char* mask = (const unsigned char*)d_in[3];
  const float* Wq = (const float*)d_in[4];
  const float* Wk = (const float*)d_in[5];
  const float* Wv = (const float*)d_in[6];
  const float* Wo = (const float*)d_in[7];

  const size_t BLD = (size_t)BL_ * D_;   // 4 Mi elements
  const size_t DD  = (size_t)D_ * D_;    // 1 Mi elements
  bf16_t* ws = (bf16_t*)d_ws;
  bf16_t* Xq = ws;           bf16_t* Xk = Xq + BLD;  bf16_t* Xv = Xk + BLD;
  bf16_t* Bq = Xv + BLD;     bf16_t* Bk = Bq + DD;
  bf16_t* Bv = Bk + DD;      bf16_t* Bo = Bv + DD;
  bf16_t* Qh = Bo + DD;      bf16_t* Kh = Qh + BLD;  bf16_t* Vt = Kh + BLD;
  bf16_t* Ct = Vt + BLD;     // 64 MB of bf16 workspace so far
  float*  invRow = (float*)(Ct + BLD);   // + B*L floats (16 KB)

  float* outMain = (float*)d_out;
  float* outTop  = outMain + BLD;        // (B, L, L) head-0 attention

  // 1) f32 -> bf16
  k_cvt<<<1024, 256, 0, stream>>>(q,  Xq, (int)BLD);
  k_cvt<<<1024, 256, 0, stream>>>(k,  Xk, (int)BLD);
  k_cvt<<<1024, 256, 0, stream>>>(v,  Xv, (int)BLD);
  k_cvt<<<256,  256, 0, stream>>>(Wq, Bq, (int)DD);
  k_cvt<<<256,  256, 0, stream>>>(Wk, Bk, (int)DD);
  k_cvt<<<256,  256, 0, stream>>>(Wv, Bv, (int)DD);
  k_cvt<<<256,  256, 0, stream>>>(Wo, Bo, (int)DD);

  // 2) projections (4096 wave-tiles -> 512 blocks x 8 waves)
  k_gemm_xt<1><<<512, 256, 0, stream>>>(Xq, Bq, nullptr, Qh, 0.125f); // 1/sqrt(DH)
  k_gemm_xt<1><<<512, 256, 0, stream>>>(Xk, Bk, nullptr, Kh, 1.0f);
  k_gemm_xt<2><<<512, 256, 0, stream>>>(Xv, Bv, nullptr, Vt, 1.0f);   // V^T layout

  // 3) attention: B*H*8 blocks of 8 waves, 2 q-tiles per wave
  k_attn<<<256, 256, 0, stream>>>(Qh, Kh, Vt, mask, Ct, outTop, invRow);

  // 3b) normalize head-0 attention rows
  k_fix<<<2048, 256, 0, stream>>>(outTop, invRow, (int)((size_t)B_ * L_ * L_));

  // 4) output projection: ctx @ Wo^T -> f32
  k_gemm_xt<0><<<512, 256, 0, stream>>>(Ct, Bo, outMain, nullptr, 1.0f);
}
